// F8Linear_23424751633180
// MI455X (gfx1250) — 8-bit matrix operand order repaired and checked by reading, compiled, not run on hardware
//
#include <hip/hip_runtime.h>
#include <hip/hip_fp16.h>

#ifndef __has_builtin
#define __has_builtin(x) 0
#endif

#define M_DIM 2048
#define N_DIM 4096
#define K_DIM 4096

#define BN 64
#define BK 128

#define E4M3_MAX 448.0f
#define E5M2_MAX 57344.0f
#define EPS_F 1e-12f

typedef __attribute__((ext_vector_type(16))) int   v16i;
typedef __attribute__((ext_vector_type(8)))  float v8f;

#if defined(__HIP_DEVICE_COMPILE__)
#  if __has_builtin(__builtin_amdgcn_wmma_f32_16x16x128_bf8_fp8)
#    define WMMA_A_BF8_B_FP8(A, B, C) \
       __builtin_amdgcn_wmma_f32_16x16x128_bf8_fp8((A), (B), (short)0, (C), false, false)
#  elif __has_builtin(__builtin_amdgcn_wmma_f32_16x16x128_fp8_fp8)
#    define WMMA_A_BF8_B_FP8(A, B, C) \
       __builtin_amdgcn_wmma_f32_16x16x128_fp8_fp8((A), (B), (short)0, (C), false, false)
#  else
#    error "device pass: no 16x16x128 fp8 wmma builtin available"
#  endif
#  if __has_builtin(__builtin_amdgcn_cvt_pk_bf8_f32) && __has_builtin(__builtin_amdgcn_cvt_pk_fp8_f32)
#    define HW_CVT 1
#  else
#    define HW_CVT 0
#  endif
#  if __has_builtin(__builtin_amdgcn_global_load_async_to_lds_b128)
#    define HAVE_ASYNC 1
#  else
#    define HAVE_ASYNC 0
#  endif
#else
#  define WMMA_A_BF8_B_FP8(A, B, C) (C)
#  define HW_CVT 0
#  define HAVE_ASYNC 0
#endif

#if HAVE_ASYNC
typedef int v4i32 __attribute__((__vector_size__(4 * sizeof(int))));
typedef __attribute__((address_space(1))) v4i32 as1_v4i;
typedef __attribute__((address_space(3))) v4i32 as3_v4i;
__device__ inline void async_wait_all() {
#  if __has_builtin(__builtin_amdgcn_s_wait_asynccnt)
  __builtin_amdgcn_s_wait_asynccnt(0);
#  else
  asm volatile("s_wait_asynccnt 0x0" ::: "memory");
#  endif
}
#endif

__device__ inline unsigned int f32_to_e5m2_sw(float f) {
  f = fminf(fmaxf(f, -E5M2_MAX), E5M2_MAX);
  unsigned short h = __half_as_ushort(__float2half(f));
  unsigned short lsb = (unsigned short)((h >> 8) & 1);
  h = (unsigned short)(h + 0x7F + lsb);
  return (unsigned int)(h >> 8) & 0xFFu;
}

__device__ inline unsigned int f32_to_e4m3_sw(float f) {
  f = fminf(fmaxf(f, -E4M3_MAX), E4M3_MAX);
  unsigned int u = __float_as_uint(f);
  unsigned int sign = (u >> 31) << 7;
  unsigned int exp  = (u >> 23) & 0xFF;
  unsigned int man  = u & 0x7FFFFF;
  if (exp == 0) return sign;
  int e = (int)exp - 127;
  if (e < -9) return sign;
  if (e < -6) {
    int shift = -6 - e;
    unsigned int mant = man | 0x800000u;
    int rshift = 20 + shift;
    unsigned int keep = mant >> rshift;
    unsigned int rem  = mant & ((1u << rshift) - 1u);
    unsigned int half = 1u << (rshift - 1);
    keep += (rem > half) || (rem == half && (keep & 1u));
    return sign | keep;
  }
  unsigned int m3  = man >> 20;
  unsigned int rem = man & 0xFFFFFu;
  m3 += (rem > 0x80000u) || (rem == 0x80000u && (m3 & 1u));
  unsigned int eb = (unsigned int)(e + 7);
  if (m3 == 8u) { m3 = 0u; eb += 1u; }
  if (eb > 15u || (eb == 15u && m3 > 6u)) { eb = 15u; m3 = 6u; }
  return sign | (eb << 3) | m3;
}


__global__ void init_amax(unsigned int* a) {
  a[0] = 0u;
  a[1] = 0u;
}

__global__ __launch_bounds__(256) void amax_kernel(const float* __restrict__ p, int n4,
                                                   unsigned int* __restrict__ out) {
  __shared__ float red[256];
  float m = 0.0f;
  const float4* p4 = (const float4*)p;
  for (int i = blockIdx.x * blockDim.x + threadIdx.x; i < n4; i += gridDim.x * blockDim.x) {
    float4 v = p4[i];
    m = fmaxf(m, fabsf(v.x));
    m = fmaxf(m, fabsf(v.y));
    m = fmaxf(m, fabsf(v.z));
    m = fmaxf(m, fabsf(v.w));
  }
  red[threadIdx.x] = m;
  __syncthreads();
  for (int s = 128; s > 0; s >>= 1) {
    if ((int)threadIdx.x < s) red[threadIdx.x] = fmaxf(red[threadIdx.x], red[threadIdx.x + s]);
    __syncthreads();
  }
  if (threadIdx.x == 0) atomicMax(out, __float_as_uint(red[0]));
}

__global__ __launch_bounds__(256) void quant_e5m2(const float* __restrict__ in,
                                                  const unsigned int* __restrict__ amax,
                                                  unsigned int* __restrict__ outq, int n4) {
  int i = blockIdx.x * blockDim.x + threadIdx.x;
  if (i >= n4) return;
  const float a = fmaxf(__uint_as_float(amax[0]), EPS_F);
  const float s = E5M2_MAX / a;
  const float4 v = ((const float4*)in)[i];
  float x0 = fminf(fmaxf(v.x * s, -E5M2_MAX), E5M2_MAX);
  float x1 = fminf(fmaxf(v.y * s, -E5M2_MAX), E5M2_MAX);
  float x2 = fminf(fmaxf(v.z * s, -E5M2_MAX), E5M2_MAX);
  float x3 = fminf(fmaxf(v.w * s, -E5M2_MAX), E5M2_MAX);
#if HW_CVT
  int p = __builtin_amdgcn_cvt_pk_bf8_f32(x0, x1, 0, false);
  p = __builtin_amdgcn_cvt_pk_bf8_f32(x2, x3, p, true);
  outq[i] = (unsigned int)p;
#else
  outq[i] = f32_to_e5m2_sw(x0) | (f32_to_e5m2_sw(x1) << 8) |
            (f32_to_e5m2_sw(x2) << 16) | (f32_to_e5m2_sw(x3) << 24);
#endif
}

__global__ __launch_bounds__(256) void quant_e4m3(const float* __restrict__ in,
                                                  const unsigned int* __restrict__ amax,
                                                  unsigned int* __restrict__ outq, int n4) {
  int i = blockIdx.x * blockDim.x + threadIdx.x;
  if (i >= n4) return;
  const float a = fmaxf(__uint_as_float(amax[0]), EPS_F);
  const float s = E4M3_MAX / a;
  const float4 v = ((const float4*)in)[i];
  float x0 = fminf(fmaxf(v.x * s, -E4M3_MAX), E4M3_MAX);
  float x1 = fminf(fmaxf(v.y * s, -E4M3_MAX), E4M3_MAX);
  float x2 = fminf(fmaxf(v.z * s, -E4M3_MAX), E4M3_MAX);
  float x3 = fminf(fmaxf(v.w * s, -E4M3_MAX), E4M3_MAX);
#if HW_CVT
  int p = __builtin_amdgcn_cvt_pk_fp8_f32(x0, x1, 0, false);
  p = __builtin_amdgcn_cvt_pk_fp8_f32(x2, x3, p, true);
  outq[i] = (unsigned int)p;
#else
  outq[i] = f32_to_e4m3_sw(x0) | (f32_to_e4m3_sw(x1) << 8) |
            (f32_to_e4m3_sw(x2) << 16) | (f32_to_e4m3_sw(x3) << 24);
#endif
}

__device__ inline void stage_b(const unsigned char* __restrict__ gbase,
                               unsigned char* __restrict__ lbase, int tid) {
#pragma unroll
  for (int j = 0; j < 2; ++j) {
    const int c   = tid + j * 256;
    const int row = c >> 3;
    const int ko  = (c & 7) * 16;
    const unsigned char* g = gbase + (size_t)row * K_DIM + ko;
    unsigned char* l = lbase + row * BK + ko;
#if HAVE_ASYNC
    __builtin_amdgcn_global_load_async_to_lds_b128((as1_v4i*)g, (as3_v4i*)l, 0, 0);
#else
    const int4 v = *(const int4*)g;
    *(int4*)l = v;
#endif
  }
}

__global__ __launch_bounds__(256) void f8_gemm(const unsigned char* __restrict__ xq,
                                               const unsigned char* __restrict__ wq,
                                               const unsigned int* __restrict__ amax,
                                               const float* __restrict__ bias,
                                               float* __restrict__ out) {
  __shared__ unsigned char bsh[2][BN * BK];

  const int tid  = threadIdx.x;
  const int lane = tid & 31;
  const int wave = tid >> 5;
  const int hi   = lane >> 4;
  const int l15  = lane & 15;
  const int mbase = blockIdx.y * 128 + wave * 16;
  const int nbase = blockIdx.x * BN;

  v8f acc[4];
#pragma unroll
  for (int t = 0; t < 4; ++t) acc[t] = (v8f){0.f, 0.f, 0.f, 0.f, 0.f, 0.f, 0.f, 0.f};

  const unsigned char* arow = xq + (size_t)(mbase + l15) * K_DIM + (size_t)hi * 16;
  const unsigned char* wtile = wq + (size_t)nbase * K_DIM;

  stage_b(wtile, &bsh[0][0], tid);
#if HAVE_ASYNC
  async_wait_all();
#endif
  __syncthreads();

  int buf = 0;
  for (int k0 = 0; k0 < K_DIM; k0 += BK) {
    if (k0 + BK < K_DIM) {
      stage_b(wtile + k0 + BK, &bsh[buf ^ 1][0], tid);
      __builtin_prefetch(arow + k0 + BK, 0, 1);
    }

    v16i a;
#pragma unroll
    for (int c = 0; c < 4; ++c) {
      const int4 t4 = *(const int4*)(arow + k0 + c * 32);
      a[4 * c]     = t4.x;
      a[4 * c + 1] = t4.y;
      a[4 * c + 2] = t4.z;
      a[4 * c + 3] = t4.w;
    }

#pragma unroll
    for (int t = 0; t < 4; ++t) {
      const unsigned char* bl = &bsh[buf][0] + (t * 16 + l15) * BK + hi * 16;
      v16i b;
#pragma unroll
      for (int c = 0; c < 4; ++c) {
        const int4 q = *(const int4*)(bl + c * 32);
        b[4 * c]     = q.x;
        b[4 * c + 1] = q.y;
        b[4 * c + 2] = q.z;
        b[4 * c + 3] = q.w;
      }
      acc[t] = WMMA_A_BF8_B_FP8(a, b, acc[t]);
    }

#if HAVE_ASYNC
    async_wait_all();
#endif
    __syncthreads();
    buf ^= 1;
  }

  const float ax = fmaxf(__uint_as_float(amax[0]), EPS_F);
  const float aw = fmaxf(__uint_as_float(amax[1]), EPS_F);
  const float dq = (ax / E5M2_MAX) * (aw / E4M3_MAX);

#pragma unroll
  for (int t = 0; t < 4; ++t) {
    const int col = nbase + t * 16 + l15;
    const float bv = bias[col];
#pragma unroll
    for (int r = 0; r < 8; ++r) {
      const int row = mbase + hi * 8 + r;
      out[(size_t)row * N_DIM + col] = acc[t][r] * dq + bv;
    }
  }
}

extern "C" void kernel_launch(void* const* d_in, const int* in_sizes, int n_in,
                              void* d_out, int out_size, void* d_ws, size_t ws_size,
                              hipStream_t stream) {
  const float* x    = (const float*)d_in[0];
  const float* w    = (const float*)d_in[1];
  const float* bias = (const float*)d_in[2];
  float* out = (float*)d_out;

  unsigned char* ws = (unsigned char*)d_ws;
  unsigned int* amax = (unsigned int*)ws;
  unsigned char* xq = ws + 256;
  unsigned char* wq = xq + (size_t)M_DIM * K_DIM;

  const int nx = M_DIM * K_DIM;
  const int nw = N_DIM * K_DIM;

  init_amax<<<1, 1, 0, stream>>>(amax);
  amax_kernel<<<512, 256, 0, stream>>>(x, nx / 4, amax + 0);
  amax_kernel<<<512, 256, 0, stream>>>(w, nw / 4, amax + 1);

  quant_e5m2<<<(nx / 4 + 255) / 256, 256, 0, stream>>>(x, amax + 0, (unsigned int*)xq, nx / 4);
  quant_e4m3<<<(nw / 4 + 255) / 256, 256, 0, stream>>>(w, amax + 1, (unsigned int*)wq, nw / 4);

  dim3 grid(N_DIM / 64, M_DIM / 128);
  f8_gemm<<<grid, 256, 0, stream>>>(xq, wq, amax, bias, out);
}
